// MultiSensor_TSCMamba_v1_63806034149887
// MI455X (gfx1250) — compile-verified, matrix operand order checked by reading, not run on hardware
//
#include <hip/hip_runtime.h>

// ---------------------------------------------------------------------------
// MultiSensor TSCMamba forward for gfx1250 (CDNA5, wave32, WMMA).
// Heavy stages (patch conv, projection, cross conv) run as bf16 WMMA GEMMs
// (v_wmma_f32_16x16x32_bf16). Kernel B stages its input tile with CDNA5
// async-to-LDS DMA (global_load_async_to_lds_b128 + s_wait_asynccnt).
// Kernel A prefetches the next K-slab at WGP scope (global_prefetch_b8).
// Everything else is tiny (B=32) VALU tail work.
// ---------------------------------------------------------------------------

typedef __attribute__((ext_vector_type(16))) __bf16 v16bf;
typedef __attribute__((ext_vector_type(8)))  float  v8f;

__device__ inline unsigned short f2bf(float f) {
  union { float fv; unsigned int u; } c; c.fv = f;
  unsigned int u = c.u;
  u += 0x7fffu + ((u >> 16) & 1u);           // round-to-nearest-even
  return (unsigned short)(u >> 16);
}
__device__ inline unsigned pk2bf(float lo, float hi) {
  return (unsigned)f2bf(lo) | ((unsigned)f2bf(hi) << 16);
}

// Load one 16x32 bf16 WMMA fragment half-row pair: two b128 loads.
// Register mapping assumed: element e -> K = (e<8?0:16) + laneHalfOff + e%8.
__device__ inline v16bf frag_ld(const unsigned short* p) {
  union { uint4 u[2]; v16bf v; } f;
  f.u[0] = *reinterpret_cast<const uint4*>(p);
  f.u[1] = *reinterpret_cast<const uint4*>(p + 16);
  return f.v;
}

__device__ inline float geluf(float x) { return 0.5f * x * (1.f + erff(x * 0.7071067811865475f)); }
__device__ inline float siluf(float x) { return x / (1.f + expf(-x)); }
__device__ inline float softplusf(float x) { return (x > 20.f) ? x : log1pf(expf(x)); }

// ---------------------------------------------------------------------------
// Kernel W: pack weights to bf16 (conv weights are already K-contiguous) and
// zero the K-pad columns of the bf16 conv-output buffer Ybf.
// Wc: [3][32][8192]  Pw: [3][256][224] (K pad 196->224)  Cw: [256][1536]
// Ybf: [3][32][32][224] (only cols 196..223 zeroed here; rest written by A).
// ---------------------------------------------------------------------------
__global__ void k_prep(const float* __restrict__ cw0, const float* __restrict__ cw1,
                       const float* __restrict__ cw2, const float* __restrict__ pw0,
                       const float* __restrict__ pw1, const float* __restrict__ pw2,
                       const float* __restrict__ ccw,
                       unsigned short* __restrict__ Wc, unsigned short* __restrict__ Pw,
                       unsigned short* __restrict__ Cwb, unsigned short* __restrict__ Ybf) {
  int gid = blockIdx.x * 256 + threadIdx.x;
  if (gid < 786432) {
    int s = gid / 262144, rem = gid % 262144;
    const float* src = (s == 0) ? cw0 : ((s == 1) ? cw1 : cw2);
    Wc[gid] = f2bf(src[rem]);
  } else if (gid < 958464) {
    int i = gid - 786432;
    int s = i / 57344, rem = i % 57344;
    int l = rem / 224, k = rem % 224;
    const float* src = (s == 0) ? pw0 : ((s == 1) ? pw1 : pw2);
    Pw[i] = f2bf((k < 196) ? src[l * 196 + k] : 0.f);
  } else if (gid < 1351680) {
    int i = gid - 958464;
    Cwb[i] = f2bf(ccw[i]);
  } else if (gid < 1437696) {
    int i = gid - 1351680;            // 3*32*32*28 pad elements
    int sbc = i / 28, k = 196 + i % 28;
    Ybf[(size_t)sbc * 224 + k] = 0;
  }
}

// ---------------------------------------------------------------------------
// Kernel A: 16x16/s16 patch conv as WMMA GEMM.
// Block = one (sensor, batch, 16-patch tile); 2 waves = 2 co-tiles of 16.
// K = 8192 in 256 steps of 32; B slab gathered (float4-coalesced, bf16-packed)
// into LDS; A rows streamed from pre-packed bf16 weights. WGP-scope prefetch
// of the next slab.
// Out: Ybf[s][b][co=32][n pad 224] in bf16 (+bias).
// ---------------------------------------------------------------------------
__global__ void k_patch_conv(const float* __restrict__ xs0, const float* __restrict__ xs1,
                             const float* __restrict__ xs2,
                             const float* __restrict__ cb0, const float* __restrict__ cb1,
                             const float* __restrict__ cb2,
                             const unsigned short* __restrict__ Wc,
                             unsigned short* __restrict__ Ybf) {
  __shared__ __align__(16) unsigned short Bs[16][32];   // [n][k]
  int blk = blockIdx.x;
  int nt = blk % 13; int b = (blk / 13) % 32; int s = blk / (13 * 32);
  const float* xin = (s == 0) ? xs0 : ((s == 1) ? xs1 : xs2);
  const float* cb  = (s == 0) ? cb0 : ((s == 1) ? cb1 : cb2);
  int t = threadIdx.x;            // 64 threads = 2 waves
  int wave = t >> 5, lane = t & 31;
  int p = t >> 2;                 // patch slot 0..15 (staging role)
  int q = t & 3;                  // quarter of the 32-K slab
  int n = nt * 16 + p;
  bool valid = (n < 196);
  int ph = n / 14, pww = n % 14;
  const float* xb = xin + (size_t)b * 1605632;
  int coff = (lane < 16) ? 0 : 8;
  int m16  = lane & 15;
  const unsigned short* Arow = Wc + ((size_t)s * 32 + wave * 16 + m16) * 8192;
  v8f acc = {0.f, 0.f, 0.f, 0.f, 0.f, 0.f, 0.f, 0.f};

  for (int ks = 0; ks < 256; ++ks) {
    int k0 = ks * 32;
    int ci = k0 >> 8;
    int kh0 = (k0 & 255) >> 4;    // even; slab = rows kh0, kh0+1
    __syncthreads();
    {   // stage B slab: each thread loads 8 contiguous floats along kw
      int kh  = kh0 + (q >> 1);
      int kw0 = (q & 1) * 8;
      float4 v0 = {0.f,0.f,0.f,0.f}, v1 = {0.f,0.f,0.f,0.f};
      if (valid) {
        const float* src = xb + ci * 50176 + (ph * 16 + kh) * 224 + pww * 16 + kw0;
        v0 = *(const float4*)src;
        v1 = *(const float4*)(src + 4);
        // next 32-K slab (2 rows ahead); locality 3 -> WGP scope (all levels)
        __builtin_prefetch(src + 448, 0, 3);
      }
      int kk = (q >> 1) * 16 + kw0;
      unsigned* d = reinterpret_cast<unsigned*>(&Bs[p][kk]);
      d[0] = pk2bf(v0.x, v0.y);
      d[1] = pk2bf(v0.z, v0.w);
      d[2] = pk2bf(v1.x, v1.y);
      d[3] = pk2bf(v1.z, v1.w);
    }
    __syncthreads();
    v16bf a  = frag_ld(Arow + k0 + coff);
    v16bf bb = frag_ld(&Bs[m16][coff]);
    acc = __builtin_amdgcn_wmma_f32_16x16x32_bf16(false, a, false, bb, (short)0, acc,
                                                  false, false);
  }
  int nOut = nt * 16 + m16;
  if (nOut < 196) {
    int mbase = wave * 16 + ((lane < 16) ? 0 : 8);
    unsigned short* Yb16 = Ybf + ((size_t)(s * 32 + b)) * 32 * 224;
    #pragma unroll
    for (int r = 0; r < 8; ++r) {
      int co = mbase + r;
      Yb16[co * 224 + nOut] = f2bf(acc[r] + cb[co]);
    }
  }
}

// ---------------------------------------------------------------------------
// Kernel B: projection GEMM  Xr[c=32][l=256] = gelu(Y[c][196] @ pw^T + pb).
// One block per (s,b). The 14KB bf16 Y tile is staged with CDNA5 async DMA:
// global_load_async_to_lds_b128 (ASYNCcnt) -> s_wait_asynccnt 0 -> barrier.
// 8 waves x 4 tiles; K padded to 224 (7 WMMA steps).
// Out layout: Xr[s][b][l][c] (c contiguous -> cross conv friendly).
// ---------------------------------------------------------------------------
__global__ void k_proj(const unsigned short* __restrict__ Ybf,
                       const unsigned short* __restrict__ Pw,
                       const float* __restrict__ pb0, const float* __restrict__ pb1,
                       const float* __restrict__ pb2, float* __restrict__ Xr) {
  __shared__ __align__(16) unsigned short Yb[32][224];   // 14336 bytes
  int blk = blockIdx.x; int b = blk % 32, s = blk / 32;
  const float* pb  = (s == 0) ? pb0 : ((s == 1) ? pb1 : pb2);
  const unsigned short* Ysb = Ybf + ((size_t)s * 32 + b) * 32 * 224;
  int t = threadIdx.x;  // 256
  {
    unsigned ldsbase = (unsigned)(size_t)(&Yb[0][0]);
    unsigned long long gbase = (unsigned long long)(size_t)Ysb;
    for (int chunk = t; chunk < 896; chunk += 256) {   // 896 x 16B = 14336B
      unsigned la = ldsbase + (unsigned)chunk * 16u;
      unsigned long long ga = gbase + (unsigned long long)chunk * 16ull;
      asm volatile("global_load_async_to_lds_b128 %0, %1, off"
                   :: "v"(la), "v"(ga) : "memory");
    }
    asm volatile("s_wait_asynccnt 0x0" ::: "memory");
  }
  __syncthreads();
  int wave = t >> 5, lane = t & 31;
  int coff = (lane < 16) ? 0 : 8, l16 = lane & 15;
  #pragma unroll
  for (int qq = 0; qq < 4; ++qq) {
    int tile = wave * 4 + qq;
    int ctile = tile & 1, ltile = tile >> 1;
    int l = ltile * 16 + l16;
    const unsigned short* Brow = Pw + ((size_t)s * 256 + l) * 224;
    v8f acc = {0.f, 0.f, 0.f, 0.f, 0.f, 0.f, 0.f, 0.f};
    #pragma unroll
    for (int ks = 0; ks < 7; ++ks) {
      v16bf a  = frag_ld(&Yb[ctile * 16 + l16][ks * 32 + coff]);
      v16bf bb = frag_ld(Brow + ks * 32 + coff);
      acc = __builtin_amdgcn_wmma_f32_16x16x32_bf16(false, a, false, bb, (short)0, acc,
                                                    false, false);
    }
    int cbase = ctile * 16 + ((lane < 16) ? 0 : 8);
    float* dst = Xr + (((size_t)(s * 32 + b)) * 256 + l) * 32;
    float bias = pb[l];
    #pragma unroll
    for (int r = 0; r < 8; ++r) dst[cbase + r] = geluf(acc[r] + bias);
  }
}

// ---------------------------------------------------------------------------
// Kernel C: cross conv1d(k=3,pad=1) as GEMM  M=256(o) N=32(c) K=1536(i*3+t).
// One block per (pair,b); im2col gathered per 32-K step into LDS.
// Out: CR[pair][b][o][c] = gelu(bn(conv)).
// ---------------------------------------------------------------------------
__global__ void k_cross(const float* __restrict__ Xr, const unsigned short* __restrict__ Cw,
                        const float* __restrict__ ccb, const float* __restrict__ ccg,
                        const float* __restrict__ ccbe, const float* __restrict__ ccrm,
                        const float* __restrict__ ccrv, float* __restrict__ CR) {
  __shared__ __align__(16) unsigned short Bs[32][32];   // [c][k]
  int blk = blockIdx.x; int b = blk % 32, pr = blk / 32;
  int s1 = (pr == 2) ? 1 : 0;
  int s2 = (pr == 0) ? 1 : 2;
  int t = threadIdx.x, wave = t >> 5, lane = t & 31;
  int coff = (lane < 16) ? 0 : 8, l16 = lane & 15;
  int nrow = t >> 3;          // c row for staging
  int kb = (t & 7) * 4;
  v8f acc[4];
  #pragma unroll
  for (int qq = 0; qq < 4; ++qq) acc[qq] = (v8f){0.f,0.f,0.f,0.f,0.f,0.f,0.f,0.f};

  for (int ks = 0; ks < 48; ++ks) {
    __syncthreads();
    #pragma unroll
    for (int j = 0; j < 4; ++j) {
      int kk = kb + j, k = ks * 32 + kk;
      int i = k / 3, t3 = k - i * 3;
      int cpos = nrow + t3 - 1;
      float v = 0.f;
      if (cpos >= 0 && cpos < 32) {
        int sN = (i < 256) ? s1 : s2;
        int l  = (i < 256) ? i : i - 256;
        v = Xr[(((size_t)sN * 32 + b) * 256 + l) * 32 + cpos];
      }
      Bs[nrow][kk] = f2bf(v);
    }
    __syncthreads();
    #pragma unroll
    for (int qq = 0; qq < 4; ++qq) {
      int tile = wave * 4 + qq;
      int otile = tile >> 1, ntile = tile & 1;
      const unsigned short* Arow = Cw + ((size_t)(otile * 16 + l16)) * 1536;
      v16bf a  = frag_ld(Arow + ks * 32 + coff);
      v16bf bb = frag_ld(&Bs[ntile * 16 + l16][coff]);
      acc[qq] = __builtin_amdgcn_wmma_f32_16x16x32_bf16(false, a, false, bb, (short)0,
                                                        acc[qq], false, false);
    }
  }
  #pragma unroll
  for (int qq = 0; qq < 4; ++qq) {
    int tile = wave * 4 + qq;
    int otile = tile >> 1, ntile = tile & 1;
    int c = ntile * 16 + l16;
    int obase = otile * 16 + ((lane < 16) ? 0 : 8);
    float* dst = CR + (((size_t)pr * 32 + b)) * 256 * 32;
    #pragma unroll
    for (int r = 0; r < 8; ++r) {
      int o = obase + r;
      float inv = ccg[o] * rsqrtf(ccrv[o] + 1e-5f);
      float v = acc[qq][r] + ccb[o];
      v = v * inv + (ccbe[o] - ccrm[o] * inv);
      dst[o * 32 + c] = geluf(v);
    }
  }
}

// ---------------------------------------------------------------------------
// Kernel D: channel means -> feats(1536) -> fusion GEMV + BN + gelu -> LN.
// One block per batch element.
// ---------------------------------------------------------------------------
__global__ void k_fuse(const float* __restrict__ Xr, const float* __restrict__ CR,
                       const float* __restrict__ fuw, const float* __restrict__ fub,
                       const float* __restrict__ fug, const float* __restrict__ fube,
                       const float* __restrict__ furm, const float* __restrict__ furv,
                       const float* __restrict__ lng, const float* __restrict__ lnb,
                       float* __restrict__ xf) {
  __shared__ float feats[1536];
  __shared__ float red[256];
  int b = blockIdx.x, t = threadIdx.x;   // 256
  for (int g6 = 0; g6 < 6; ++g6) {
    const float* src = (g6 < 3) ? (Xr + (((size_t)g6 * 32 + b) * 256 + t) * 32)
                                : (CR + (((size_t)(g6 - 3) * 32 + b) * 256 + t) * 32);
    float s = 0.f;
    #pragma unroll
    for (int c = 0; c < 32; ++c) s += src[c];
    feats[g6 * 256 + t] = s * (1.f / 32.f);
  }
  __syncthreads();
  float a = fub[t];
  const float* wrow = fuw + (size_t)t * 1536;
  for (int k = 0; k < 1536; ++k) a += wrow[k] * feats[k];
  float inv = fug[t] * rsqrtf(furv[t] + 1e-5f);
  a = geluf(a * inv + (fube[t] - furm[t] * inv));
  red[t] = a; __syncthreads();
  for (int s = 128; s > 0; s >>= 1) { if (t < s) red[t] += red[t + s]; __syncthreads(); }
  float mu = red[0] / 256.f;
  __syncthreads();
  float dev = a - mu;
  red[t] = dev * dev; __syncthreads();
  for (int s = 128; s > 0; s >>= 1) { if (t < s) red[t] += red[t + s]; __syncthreads(); }
  float var = red[0] / 256.f;
  xf[b * 256 + t] = dev * rsqrtf(var + 1e-5f) * lng[t] + lnb[t];
}

// ---------------------------------------------------------------------------
// Kernel E: m1 stack (L=1 => scan collapses; x1 = 2*stack(xf)). Block per b.
// ---------------------------------------------------------------------------
__global__ void k_m1(const float* __restrict__ xf, const float* __restrict__ Win,
                     const float* __restrict__ Cwv, const float* __restrict__ Cbv,
                     const float* __restrict__ Wx, const float* __restrict__ Wdt,
                     const float* __restrict__ bdt, const float* __restrict__ Alog,
                     const float* __restrict__ Dp, const float* __restrict__ Wout,
                     float* __restrict__ x1) {
  __shared__ float xin[256], xcs[512], zs[512], dbc[48], ys[512];
  __shared__ float Ssh;
  (void)Alog;  // exp(dt*A)*h0 vanishes for h0=0, L=1
  int b = blockIdx.x, t = threadIdx.x;   // 512
  if (t < 256) xin[t] = xf[b * 256 + t];
  __syncthreads();
  for (int layer = 0; layer < 2; ++layer) {
    const float* Wi = Win + (size_t)layer * 1024 * 256;
    {
      float a0 = 0.f, a1 = 0.f;
      const float* r0 = Wi + (size_t)t * 256;
      const float* r1 = Wi + (size_t)(512 + t) * 256;
      for (int k = 0; k < 256; ++k) { a0 += r0[k] * xin[k]; a1 += r1[k] * xin[k]; }
      float xconv = Cbv[layer * 512 + t] + a0 * Cwv[(layer * 512 + t) * 4 + 3];
      xcs[t] = siluf(xconv);
      zs[t] = a1;
    }
    __syncthreads();
    if (t < 48) {
      float a = 0.f;
      const float* r = Wx + ((size_t)layer * 48 + t) * 512;
      for (int k = 0; k < 512; ++k) a += r[k] * xcs[k];
      dbc[t] = a;
    }
    __syncthreads();
    if (t == 0) {
      float S = 0.f;
      for (int n = 0; n < 16; ++n) S += dbc[16 + n] * dbc[32 + n];
      Ssh = S;
    }
    __syncthreads();
    {
      const float* r = Wdt + ((size_t)layer * 512 + t) * 16;
      float a = bdt[layer * 512 + t];
      #pragma unroll
      for (int rr = 0; rr < 16; ++rr) a += r[rr] * dbc[rr];
      float dtv = softplusf(a);
      float y = dtv * xcs[t] * Ssh + Dp[layer * 512 + t] * xcs[t];
      ys[t] = y * siluf(zs[t]);
    }
    __syncthreads();
    if (t < 256) {
      float a = 0.f;
      const float* r = Wout + ((size_t)layer * 256 + t) * 512;
      for (int k = 0; k < 512; ++k) a += r[k] * ys[k];
      xin[t] += a;
    }
    __syncthreads();
  }
  if (t < 256) x1[b * 256 + t] = 2.f * xin[t];
}

// ---------------------------------------------------------------------------
// Kernel F: m2 stack, L=256, Di=2, N=16. One wave per (b, direction);
// sequential scan with lanes = (d,n) states, shfl_xor reduce over n.
// ---------------------------------------------------------------------------
__global__ void k_m2(const float* __restrict__ x1, const float* __restrict__ Win2,
                     const float* __restrict__ Cw2, const float* __restrict__ Cb2,
                     const float* __restrict__ Wx2, const float* __restrict__ Wdt2,
                     const float* __restrict__ bdt2, const float* __restrict__ Alog2,
                     const float* __restrict__ Dp2, const float* __restrict__ Wout2,
                     float* __restrict__ x2a, float* __restrict__ x2b) {
  __shared__ float seq[256];
  __shared__ float xcraw[256][2], xcs[256][2], zz[256][2], dts[256][2];
  __shared__ float Bm[256][16], Cm[256][16], yv[256][2];
  int blk = blockIdx.x; int b = blk >> 1, dir = blk & 1;
  int t = threadIdx.x;   // 32
  for (int l = t; l < 256; l += 32)
    seq[l] = dir ? x1[b * 256 + 255 - l] : x1[b * 256 + l];
  __syncthreads();
  for (int layer = 0; layer < 2; ++layer) {
    const float* wi = Win2 + layer * 4;
    for (int l = t; l < 256; l += 32) {
      float x = seq[l];
      xcraw[l][0] = x * wi[0]; xcraw[l][1] = x * wi[1];
      zz[l][0] = x * wi[2];    zz[l][1] = x * wi[3];
    }
    __syncthreads();
    const float* wx = Wx2 + layer * 66;
    for (int l = t; l < 256; l += 32) {
      #pragma unroll
      for (int d = 0; d < 2; ++d) {
        float a = Cb2[layer * 2 + d];
        const float* cw = Cw2 + (layer * 2 + d) * 4;
        #pragma unroll
        for (int k = 0; k < 4; ++k) { int idx = l + k - 3; if (idx >= 0) a += xcraw[idx][d] * cw[k]; }
        xcs[l][d] = siluf(a);
      }
      float x0 = xcs[l][0], x1v = xcs[l][1];
      float d0 = x0 * wx[0] + x1v * wx[1];
      #pragma unroll
      for (int d = 0; d < 2; ++d)
        dts[l][d] = softplusf(d0 * Wdt2[layer * 2 + d] + bdt2[layer * 2 + d]);
      #pragma unroll
      for (int n = 0; n < 16; ++n) {
        Bm[l][n] = x0 * wx[(1 + n) * 2]  + x1v * wx[(1 + n) * 2 + 1];
        Cm[l][n] = x0 * wx[(17 + n) * 2] + x1v * wx[(17 + n) * 2 + 1];
      }
    }
    __syncthreads();
    {
      int d = t >> 4, n = t & 15;
      float A = -expf(Alog2[(layer * 2 + d) * 16 + n]);
      float h = 0.f;
      for (int l = 0; l < 256; ++l) {
        float dtv = dts[l][d];
        h = expf(dtv * A) * h + dtv * Bm[l][n] * xcs[l][d];
        float c = h * Cm[l][n];
        c += __shfl_xor(c, 1, 32);
        c += __shfl_xor(c, 2, 32);
        c += __shfl_xor(c, 4, 32);
        c += __shfl_xor(c, 8, 32);
        if (n == 0) yv[l][d] = c;
      }
    }
    __syncthreads();
    for (int l = t; l < 256; l += 32) {
      float o = 0.f;
      #pragma unroll
      for (int d = 0; d < 2; ++d) {
        float y = yv[l][d] + Dp2[layer * 2 + d] * xcs[l][d];
        y *= siluf(zz[l][d]);
        o += y * Wout2[layer * 2 + d];
      }
      seq[l] += o;
    }
    __syncthreads();
  }
  float* dst = dir ? x2b : x2a;
  for (int l = t; l < 256; l += 32) dst[b * 256 + l] = seq[l];
}

// ---------------------------------------------------------------------------
// Kernel G: head. x3 = x1 + x2a + x2b (ref adds the reversed branch un-flipped
// because the flip is over the singleton axis); two small GEMVs.
// ---------------------------------------------------------------------------
__global__ void k_head(const float* __restrict__ x1, const float* __restrict__ x2a,
                       const float* __restrict__ x2b, const float* __restrict__ cl1w,
                       const float* __restrict__ cl1b, const float* __restrict__ cl2w,
                       const float* __restrict__ cl2b, float* __restrict__ out) {
  __shared__ float x3[256], hh[128];
  int b = blockIdx.x, t = threadIdx.x;   // 256
  x3[t] = x1[b * 256 + t] + x2a[b * 256 + t] + x2b[b * 256 + t];
  __syncthreads();
  if (t < 128) {
    float a = cl1b[t];
    const float* r = cl1w + (size_t)t * 256;
    for (int k = 0; k < 256; ++k) a += r[k] * x3[k];
    hh[t] = a;
  }
  __syncthreads();
  if (t < 18) {
    float a = cl2b[t];
    const float* r = cl2w + (size_t)t * 128;
    for (int k = 0; k < 128; ++k) a += r[k] * hh[k];
    out[b * 18 + t] = a;
  }
}

// ---------------------------------------------------------------------------
extern "C" void kernel_launch(void* const* d_in, const int* in_sizes, int n_in,
                              void* d_out, int out_size, void* d_ws, size_t ws_size,
                              hipStream_t stream) {
  (void)in_sizes; (void)n_in; (void)out_size; (void)ws_size;
  const float* x0  = (const float*)d_in[0];
  const float* x1i = (const float*)d_in[1];
  const float* x2i = (const float*)d_in[2];
  const float* xcw = (const float*)d_in[3];  const float* xcb = (const float*)d_in[4];
  const float* xpw = (const float*)d_in[5];  const float* xpb = (const float*)d_in[6];
  const float* ycw = (const float*)d_in[7];  const float* ycb = (const float*)d_in[8];
  const float* ypw = (const float*)d_in[9];  const float* ypb = (const float*)d_in[10];
  const float* zcw = (const float*)d_in[11]; const float* zcb = (const float*)d_in[12];
  const float* zpw = (const float*)d_in[13]; const float* zpb = (const float*)d_in[14];
  const float* ccw = (const float*)d_in[15]; const float* ccb  = (const float*)d_in[16];
  const float* ccg = (const float*)d_in[17]; const float* ccbe = (const float*)d_in[18];
  const float* ccrm = (const float*)d_in[19]; const float* ccrv = (const float*)d_in[20];
  const float* fuw = (const float*)d_in[21]; const float* fub  = (const float*)d_in[22];
  const float* fug = (const float*)d_in[23]; const float* fube = (const float*)d_in[24];
  const float* furm = (const float*)d_in[25]; const float* furv = (const float*)d_in[26];
  const float* lng = (const float*)d_in[27]; const float* lnb = (const float*)d_in[28];
  const float* m1in = (const float*)d_in[29]; const float* m1cw = (const float*)d_in[30];
  const float* m1cb = (const float*)d_in[31]; const float* m1xw = (const float*)d_in[32];
  const float* m1dtw = (const float*)d_in[33]; const float* m1dtb = (const float*)d_in[34];
  const float* m1al = (const float*)d_in[35]; const float* m1d = (const float*)d_in[36];
  const float* m1ow = (const float*)d_in[37];
  const float* m2in = (const float*)d_in[38]; const float* m2cw = (const float*)d_in[39];
  const float* m2cb = (const float*)d_in[40]; const float* m2xw = (const float*)d_in[41];
  const float* m2dtw = (const float*)d_in[42]; const float* m2dtb = (const float*)d_in[43];
  const float* m2al = (const float*)d_in[44]; const float* m2d = (const float*)d_in[45];
  const float* m2ow = (const float*)d_in[46];
  const float* cl1w = (const float*)d_in[47]; const float* cl1b = (const float*)d_in[48];
  const float* cl2w = (const float*)d_in[49]; const float* cl2b = (const float*)d_in[50];
  float* out = (float*)d_out;

  // Workspace layout (~10.5 MB total)
  unsigned short* Ybf = (unsigned short*)d_ws;      // 3*32*32*224 bf16 = 688128
  float* Xr   = (float*)(Ybf + 688128);             // 3*32*256*32
  float* CR   = Xr + 786432;                        // 3*32*256*32
  float* xfw  = CR + 786432;                        // 32*256
  float* x1w  = xfw + 8192;
  float* x2aw = x1w + 8192;
  float* x2bw = x2aw + 8192;
  unsigned short* Wc  = (unsigned short*)(x2bw + 8192);  // 3*32*8192 bf16
  unsigned short* Pw  = Wc + 786432;                     // 3*256*224 bf16
  unsigned short* Cwb = Pw + 172032;                     // 256*1536 bf16

  k_prep      <<<5616, 256, 0, stream>>>(xcw, ycw, zcw, xpw, ypw, zpw, ccw, Wc, Pw, Cwb, Ybf);
  k_patch_conv<<<1248,  64, 0, stream>>>(x0, x1i, x2i, xcb, ycb, zcb, Wc, Ybf);
  k_proj      <<<  96, 256, 0, stream>>>(Ybf, Pw, xpb, ypb, zpb, Xr);
  k_cross     <<<  96, 256, 0, stream>>>(Xr, Cwb, ccb, ccg, ccbe, ccrm, ccrv, CR);
  k_fuse      <<<  32, 256, 0, stream>>>(Xr, CR, fuw, fub, fug, fube, furm, furv, lng, lnb, xfw);
  k_m1        <<<  32, 512, 0, stream>>>(xfw, m1in, m1cw, m1cb, m1xw, m1dtw, m1dtb, m1al, m1d, m1ow, x1w);
  k_m2        <<<  64,  32, 0, stream>>>(x1w, m2in, m2cw, m2cb, m2xw, m2dtw, m2dtb, m2al, m2d, m2ow, x2aw, x2bw);
  k_head      <<<  32, 256, 0, stream>>>(x1w, x2aw, x2bw, cl1w, cl1b, cl2w, cl2b, out);
}
